// LigerFusedLinearKLTopKLogprobLoss_80642305950034
// MI455X (gfx1250) — matrix operand order repaired and checked by reading, not compiled, not run on hardware
//
#include <hip/hip_runtime.h>

// Problem constants (from reference setup_inputs)
#define BM 128
#define BN 128
#define BK 64
#define D_DIM 2048
#define V_DIM 32768
#define M_DIM 4096
#define K_TOP 64
#define N_SEQ 2048
#define NT (V_DIM / BN)   // 256 column tiles
#define NKC (D_DIM / BK)  // 32 k-chunks
#define ASTR (BK + 8)     // padded LDS row stride in ushorts (72) -> conflict-free frag loads
#define CSTR BN           // C tile stride (staggered access avoids conflicts)

typedef __attribute__((ext_vector_type(16))) __bf16 v16bf;
typedef __attribute__((ext_vector_type(2)))  __bf16 v2bf;
typedef __attribute__((ext_vector_type(8)))  float v8f;
typedef __attribute__((ext_vector_type(4)))  float f32x4;
typedef __attribute__((ext_vector_type(4)))  unsigned int u32x4;
typedef __attribute__((ext_vector_type(2)))  unsigned int u32x2;

union FragAB { v16bf v; u32x4 q[2]; };

// packed fp32x2 -> bf16x2: one VALU op (v_cvt_pk_bf16_f32) on CDNA5.
// clang-22 does not declare the builtin for gfx1250, so emit the ISA op directly.
static __device__ __forceinline__ unsigned int pk2bf(float lo, float hi) {
#if __has_builtin(__builtin_amdgcn_cvt_pk_bf16_f32)
  v2bf t = __builtin_amdgcn_cvt_pk_bf16_f32(lo, hi);
  return __builtin_bit_cast(unsigned int, t);
#else
  unsigned int r;
  asm("v_cvt_pk_bf16_f32 %0, %1, %2" : "=v"(r) : "v"(lo), "v"(hi));
  return r;
#endif
}

static __device__ __forceinline__ float wave_max(float v) {
  #pragma unroll
  for (int o = 16; o; o >>= 1) v = fmaxf(v, __shfl_xor(v, o, 32));
  return v;
}
static __device__ __forceinline__ float wave_sum(float v) {
  #pragma unroll
  for (int o = 16; o; o >>= 1) v += __shfl_xor(v, o, 32);
  return v;
}

// ---------------------------------------------------------------------------
// Kernel 1: fused GEMM (bf16 WMMA) + per-column-tile online-softmax partials
//           + top-K / gold-label logit gather.
// grid = (M/BM, V/BN); blockIdx.x fastest so 32 consecutive blocks share one
// W column tile (1 MB) in L2; x (32 MB) lives in L2 for the whole pass.
// ---------------------------------------------------------------------------
__global__ __launch_bounds__(256) void fused_gemm_topk(
    const float* __restrict__ X, const float* __restrict__ W,
    const int* __restrict__ tk, const int* __restrict__ labels,
    float* __restrict__ pmax, float* __restrict__ psum,
    float* __restrict__ gath)
{
  __shared__ union {
    struct { unsigned short A[2][BM * ASTR]; unsigned short B[2][BN * ASTR]; } g; // 72 KB
    float C[BM * CSTR];                                                           // 64 KB
  } sm;

  const int tid  = threadIdx.x;
  const int lane = tid & 31;
  const int wave = tid >> 5;
  const int wm   = wave >> 2;   // 0..1  (64-row slab)
  const int wn   = wave & 3;    // 0..3  (32-col slab)
  const int m0   = blockIdx.x * BM;
  const int n0   = blockIdx.y * BN;

  const v8f vzero = {0.f, 0.f, 0.f, 0.f, 0.f, 0.f, 0.f, 0.f};
  v8f acc[4][2];
  #pragma unroll
  for (int i = 0; i < 4; ++i)
    #pragma unroll
    for (int j = 0; j < 2; ++j) acc[i][j] = vzero;

  f32x4 av[8], bv[8];

  // global fp32 -> regs (8 float4 each for A and B tile per thread)
  auto loadRegs = [&](int c) {
    const int kc = c * BK;
    #pragma unroll
    for (int i = 0; i < 8; ++i) {
      const int f = tid + i * 256;      // 0..2047 float4 slots
      const int r = f >> 4;             // row 0..127
      const int col = (f & 15) << 2;    // k col 0..60
      av[i] = *(const f32x4*)(X + (m0 + r) * D_DIM + kc + col);
      bv[i] = *(const f32x4*)(W + (n0 + r) * D_DIM + kc + col);
    }
  };
  // regs -> bf16 -> LDS via packed conversion
  auto cvtStore = [&](int buf) {
    #pragma unroll
    for (int i = 0; i < 8; ++i) {
      const int f = tid + i * 256;
      const int r = f >> 4;
      const int col = (f & 15) << 2;
      u32x2 ta, tb;
      ta[0] = pk2bf(av[i][0], av[i][1]); ta[1] = pk2bf(av[i][2], av[i][3]);
      tb[0] = pk2bf(bv[i][0], bv[i][1]); tb[1] = pk2bf(bv[i][2], bv[i][3]);
      *(u32x2*)&sm.g.A[buf][r * ASTR + col] = ta;
      *(u32x2*)&sm.g.B[buf][r * ASTR + col] = tb;
    }
  };
  // two 16x16x32 WMMA k-steps over the block tile (4x2 frags per wave each)
  auto computeChunk = [&](int buf) {
    const int rsub = lane & 15;
    #pragma unroll
    for (int kk = 0; kk < 2; ++kk) {
      FragAB a[4], b[2];
      const int koA = kk * 32 + ((lane >> 4) << 3); // A: lanes<16 K 0-7/16-23, lanes>=16 K 8-15/24-31
      const int koB = kk * 32 + ((lane >> 4) << 3); // B: lanes<16 K 0-7/16-23, lanes>=16 K 8-15/24-31
      #pragma unroll
      for (int i = 0; i < 4; ++i) {
        const unsigned short* p = &sm.g.A[buf][(wm * 64 + i * 16 + rsub) * ASTR + koA];
        a[i].q[0] = *(const u32x4*)p;
        a[i].q[1] = *(const u32x4*)(p + 16);
      }
      #pragma unroll
      for (int j = 0; j < 2; ++j) {
        const unsigned short* p = &sm.g.B[buf][(wn * 32 + j * 16 + rsub) * ASTR + koB];
        b[j].q[0] = *(const u32x4*)p;
        b[j].q[1] = *(const u32x4*)(p + 16);
      }
      #pragma unroll
      for (int i = 0; i < 4; ++i)
        #pragma unroll
        for (int j = 0; j < 2; ++j)
          acc[i][j] = __builtin_amdgcn_wmma_f32_16x16x32_bf16(
              false, a[i].v, false, b[j].v, (short)0, acc[i][j], false, false);
    }
  };

  // software-pipelined main loop (double-buffered LDS)
  loadRegs(0);
  cvtStore(0);
  __syncthreads();
  for (int c = 0; c < NKC; ++c) {
    if (c + 1 < NKC) loadRegs(c + 1);
    computeChunk(c & 1);
    if (c + 1 < NKC) cvtStore((c + 1) & 1);
    __syncthreads();
  }

  // ---- epilogue: C frags -> LDS (union reuse is safe after final barrier) ----
  #pragma unroll
  for (int i = 0; i < 4; ++i)
    #pragma unroll
    for (int j = 0; j < 2; ++j)
      #pragma unroll
      for (int r = 0; r < 8; ++r) {
        const int row = wm * 64 + i * 16 + ((lane >> 4) << 3) + r;
        const int col = wn * 32 + j * 16 + (lane & 15);
        sm.C[row * CSTR + col] = acc[i][j][r];
      }
  __syncthreads();

  // per-row partial max / sum-exp over this 128-column tile (staggered -> no bank conflicts)
  if (tid < BM) {
    const float* crow = &sm.C[tid * CSTR];
    float mx = -3.402823466e38f;
    #pragma unroll 8
    for (int c = 0; c < BN; ++c) mx = fmaxf(mx, crow[(c + tid) & (BN - 1)]);
    float s = 0.f;
    #pragma unroll 8
    for (int c = 0; c < BN; ++c) s += __expf(crow[(c + tid) & (BN - 1)] - mx);
    pmax[(size_t)(m0 + tid) * NT + blockIdx.y] = mx;
    psum[(size_t)(m0 + tid) * NT + blockIdx.y] = s;
  }

  // gather top-K logits (j<64) and gold-label logit (j==64) landing in this tile
  for (int i = tid; i < BM * (K_TOP + 1); i += 256) {
    const int row = i / (K_TOP + 1);
    const int j = i - row * (K_TOP + 1);
    const int m = m0 + row;
    int id;
    if (j < K_TOP) {
      id = tk[(size_t)m * K_TOP + j];
    } else {
      const int b = m / N_SEQ, n = m % N_SEQ;          // next-token shift
      id = (n + 1 < N_SEQ) ? labels[b * N_SEQ + n + 1] : -100;
    }
    const int c = id - n0;
    if (c >= 0 && c < BN) gath[(size_t)m * (K_TOP + 1) + j] = sm.C[row * CSTR + c];
  }
}

// ---------------------------------------------------------------------------
// Kernel 2: one wave per row — merge tile partials into lse, CE + top-K KL.
// With T=1 and NORMALIZE_TOPK, normalized student top-K logprob = g - lsek.
// ---------------------------------------------------------------------------
__global__ __launch_bounds__(256) void kd_ce_rows(
    const float* __restrict__ pmax, const float* __restrict__ psum,
    const float* __restrict__ gath, const float* __restrict__ tlp,
    const int* __restrict__ tmask, const int* __restrict__ labels,
    float* __restrict__ bpart)
{
  __shared__ float sred[8];
  const int lane = threadIdx.x & 31;
  const int wave = threadIdx.x >> 5;
  const int m = blockIdx.x * 8 + wave;

  // full-vocab logsumexp from 256 tile partials
  const float* pm = pmax + (size_t)m * NT;
  const float* ps = psum + (size_t)m * NT;
  float mx = -3.402823466e38f;
  #pragma unroll
  for (int t = lane; t < NT; t += 32) mx = fmaxf(mx, pm[t]);
  mx = wave_max(mx);
  float s = 0.f;
  #pragma unroll
  for (int t = lane; t < NT; t += 32) s += ps[t] * __expf(pm[t] - mx);
  s = wave_sum(s);
  const float lse = mx + __logf(s);

  // top-K normalization: lsek = logsumexp_K(gathered logits)
  const float* g = gath + (size_t)m * (K_TOP + 1);
  const float g0 = g[lane], g1 = g[lane + 32];
  const float km = wave_max(fmaxf(g0, g1));
  const float ks = wave_sum(__expf(g0 - km) + __expf(g1 - km));
  const float lsek = km + __logf(ks);

  // forward-KL terms (mask applied after normalization, matching reference)
  float kd = 0.f;
  {
    float tl = tlp[(size_t)m * K_TOP + lane];
    kd += tmask[(size_t)m * K_TOP + lane] ? __expf(tl) * (tl - (g0 - lsek)) : 0.f;
    tl = tlp[(size_t)m * K_TOP + lane + 32];
    kd += tmask[(size_t)m * K_TOP + lane + 32] ? __expf(tl) * (tl - (g1 - lsek)) : 0.f;
  }
  kd = wave_sum(kd);

  if (lane == 0) {
    const int b = m / N_SEQ, n = m % N_SEQ;
    float ce = 0.f;
    if (n + 1 < N_SEQ) {
      const int lbl = labels[b * N_SEQ + n + 1];
      if (lbl != -100) ce = lse - g[K_TOP];
    }
    sred[wave] = 0.5f * kd + 0.5f * ce;   // W_SOFT*kd + W_HARD*ce per row
  }
  __syncthreads();
  if (threadIdx.x == 0) {
    float t = 0.f;
    #pragma unroll
    for (int i = 0; i < 8; ++i) t += sred[i];
    bpart[blockIdx.x] = t;
  }
}

// ---------------------------------------------------------------------------
// Kernel 3: final scalar reduction over 512 block partials.
// ---------------------------------------------------------------------------
__global__ __launch_bounds__(512) void final_reduce(
    const float* __restrict__ bpart, float* __restrict__ out)
{
  __shared__ float smem[16];
  const int tid = threadIdx.x, lane = tid & 31, wave = tid >> 5;
  float v = wave_sum(bpart[tid]);
  if (lane == 0) smem[wave] = v;
  __syncthreads();
  if (tid < 32) {
    float x = (tid < 16) ? smem[tid] : 0.f;
    x = wave_sum(x);
    if (tid == 0) out[0] = x;
  }
}

extern "C" void kernel_launch(void* const* d_in, const int* in_sizes, int n_in,
                              void* d_out, int out_size, void* d_ws, size_t ws_size,
                              hipStream_t stream) {
  (void)in_sizes; (void)n_in; (void)out_size; (void)ws_size;
  const float* X      = (const float*)d_in[0];  // (M, D)
  const float* W      = (const float*)d_in[1];  // (V, D)
  const int*   tk     = (const int*)d_in[2];    // (M, K)
  const float* tlp    = (const float*)d_in[3];  // (M, K)
  const int*   tmask  = (const int*)d_in[4];    // (M, K)
  const int*   labels = (const int*)d_in[5];    // (B, N)
  float* out = (float*)d_out;

  float* ws    = (float*)d_ws;
  float* pmax  = ws;                                         // M*NT
  float* psum  = pmax + (size_t)M_DIM * NT;                  // M*NT
  float* gath  = psum + (size_t)M_DIM * NT;                  // M*(K+1)
  float* bpart = gath + (size_t)M_DIM * (K_TOP + 1);         // 512

  dim3 grid(M_DIM / BM, V_DIM / BN);   // (32, 256): row tiles fastest -> W-tile L2 reuse
  fused_gemm_topk<<<grid, 256, 0, stream>>>(X, W, tk, labels, pmax, psum, gath);
  kd_ce_rows<<<M_DIM / 8, 256, 0, stream>>>(pmax, psum, gath, tlp, tmask, labels, bpart);
  final_reduce<<<1, 512, 0, stream>>>(bpart, out);
}
